// DecoderBlock_54047868453440
// MI455X (gfx1250) — compile-verified, matrix operand order checked by reading, not run on hardware
//
#include <hip/hip_runtime.h>
#include <math.h>

typedef __attribute__((ext_vector_type(16))) _Float16 v16h;
typedef __attribute__((ext_vector_type(8)))  _Float16 v8h;
typedef __attribute__((ext_vector_type(8)))  float    v8f;
typedef _Float16 half_t;

#define B_  2
#define T_  2048
#define C_  1024
#define H_  16
#define DH_ 64
#define MTOK (B_*T_)   // 4096

union HV { v16h v; v8h h[2]; };

// Load one 16x32 f16 A-fragment (also used for B from a transposed [N,K] buffer).
// CDNA5 layout: lane<16 -> row=lane, k-halves {k0..k0+7, k0+16..k0+23};
//               lane>=16 -> row=lane-16, k-halves {k0+8..k0+15, k0+24..k0+31}.
__device__ inline v16h load_frag_f16(const half_t* __restrict__ A, int lda,
                                     int row0, int k0, int lane) {
  int r  = row0 + (lane & 15);
  int kb = k0 + ((lane >> 4) << 3);
  HV u;
  u.h[0] = *(const v8h*)(A + (size_t)r * lda + kb);
  u.h[1] = *(const v8h*)(A + (size_t)r * lda + kb + 16);
  return u.v;
}

// Same fragment but sourced from f32 (attention probabilities), converted in-register.
__device__ inline v16h load_frag_f32(const float* __restrict__ A, int lda,
                                     int row0, int k0, int lane) {
  int r  = row0 + (lane & 15);
  int kb = k0 + ((lane >> 4) << 3);
  const float* p = A + (size_t)r * lda + kb;
  HV u;
#pragma unroll
  for (int i = 0; i < 8; ++i) {
    u.v[i]     = (half_t)p[i];
    u.v[8 + i] = (half_t)p[16 + i];
  }
  return u.v;
}

// ---------------- generic WMMA GEMM: out = act(A[MxK]f16 @ Bt[NxK]f16^T + bias) + residual
// Wave tile: 32x64 (2 A-fragments x 4 B-fragments -> 8 WMMAs per K-step, B reused x2).
// Block: 8 waves = 2x4 -> 64x256 tile.
__global__ void __launch_bounds__(256)
wmma_gemm_bias(const half_t* __restrict__ A, const half_t* __restrict__ Bt,
               const float* __restrict__ bias, const float* __restrict__ residual,
               float* __restrict__ out32, half_t* __restrict__ out16,
               int M, int N, int K, int act)
{
  int lane = threadIdx.x & 31;
  int wave = threadIdx.x >> 5;
  int row0 = blockIdx.x * 64  + (wave >> 2) * 32;
  int col0 = blockIdx.y * 256 + (wave & 3) * 64;
  if (row0 >= M || col0 >= N) return;

  v8f acc[2][4] = {};
  for (int k0 = 0; k0 < K; k0 += 32) {
    if (k0 + 64 <= K) {  // prefetch next 128B line of each stream (global_prefetch_b8)
      __builtin_prefetch(A  + (size_t)(row0 + (lane & 31)) * K + k0 + 64, 0, 0);
      __builtin_prefetch(Bt + (size_t)(col0 + (lane & 31)) * K + k0 + 64, 0, 0);
    }
    v16h a0 = load_frag_f16(A, K, row0,      k0, lane);
    v16h a1 = load_frag_f16(A, K, row0 + 16, k0, lane);
#pragma unroll
    for (int j = 0; j < 4; ++j) {
      v16h b = load_frag_f16(Bt, K, col0 + 16 * j, k0, lane);
      acc[0][j] = __builtin_amdgcn_wmma_f32_16x16x32_f16(false, a0, false, b,
                                                         (short)0, acc[0][j], false, false);
      acc[1][j] = __builtin_amdgcn_wmma_f32_16x16x32_f16(false, a1, false, b,
                                                         (short)0, acc[1][j], false, false);
    }
  }
  int cn = lane & 15;
  int rb = (lane >> 4) << 3;   // rows 0..7 (lanes<16) or 8..15 (lanes>=16)
#pragma unroll
  for (int j = 0; j < 4; ++j) {
    int col = col0 + 16 * j + cn;
    float bv = bias ? bias[col] : 0.0f;
#pragma unroll
    for (int i = 0; i < 2; ++i) {
#pragma unroll
      for (int r = 0; r < 8; ++r) {
        int row = row0 + 16 * i + rb + r;
        float v = acc[i][j][r] + bv;
        if (act == 1) v = 0.5f * v * (1.0f + erff(v * 0.70710678118f));  // exact GELU
        if (residual) v += residual[(size_t)row * N + col];
        if (out32) out32[(size_t)row * N + col] = v;
        if (out16) out16[(size_t)row * N + col] = (half_t)v;
      }
    }
  }
}

// ---------------- attention scores: S = mask(Q K^T / 8), written into d_out att region
// Wave tile 32x64; block 8 waves = 2x4 -> 64x256 tile of S.
__global__ void __launch_bounds__(256)
attn_scores(const half_t* __restrict__ q16, const half_t* __restrict__ k16,
            float* __restrict__ att)
{
  int lane = threadIdx.x & 31;
  int wave = threadIdx.x >> 5;
  int bh    = blockIdx.z;
  int qrow0 = blockIdx.x * 64  + (wave >> 2) * 32;
  int kcol0 = blockIdx.y * 256 + (wave & 3) * 64;
  const half_t* Q  = q16 + (size_t)bh * T_ * DH_;
  const half_t* Km = k16 + (size_t)bh * T_ * DH_;
  float* S = att + (size_t)bh * T_ * T_;

  v8f acc[2][4] = {};
  if (kcol0 <= qrow0 + 31) {          // wave-uniform: skip fully masked tiles
    for (int k0 = 0; k0 < DH_; k0 += 32) {
      v16h a0 = load_frag_f16(Q, DH_, qrow0,      k0, lane);
      v16h a1 = load_frag_f16(Q, DH_, qrow0 + 16, k0, lane);
#pragma unroll
      for (int j = 0; j < 4; ++j) {
        v16h b = load_frag_f16(Km, DH_, kcol0 + 16 * j, k0, lane);
        acc[0][j] = __builtin_amdgcn_wmma_f32_16x16x32_f16(false, a0, false, b,
                                                           (short)0, acc[0][j], false, false);
        acc[1][j] = __builtin_amdgcn_wmma_f32_16x16x32_f16(false, a1, false, b,
                                                           (short)0, acc[1][j], false, false);
      }
    }
  }
  const float scale = 0.125f;  // 1/sqrt(64)
  int cn = lane & 15, rb = (lane >> 4) << 3;
#pragma unroll
  for (int j = 0; j < 4; ++j) {
    int col = kcol0 + 16 * j + cn;
#pragma unroll
    for (int i = 0; i < 2; ++i) {
#pragma unroll
      for (int r = 0; r < 8; ++r) {
        int row = qrow0 + 16 * i + rb + r;
        S[(size_t)row * T_ + col] = (col <= row) ? acc[i][j][r] * scale : -INFINITY;
      }
    }
  }
}

// ---------------- row softmax in place (masked -inf entries become exact 0)
__global__ void __launch_bounds__(256)
softmax_rows(float* __restrict__ att)
{
  __shared__ float red[256];
  float* p = att + (size_t)blockIdx.x * T_;
  int t = threadIdx.x;
  float m = -INFINITY;
  for (int i = t; i < T_; i += 256) m = fmaxf(m, p[i]);
  red[t] = m; __syncthreads();
  for (int s = 128; s > 0; s >>= 1) { if (t < s) red[t] = fmaxf(red[t], red[t + s]); __syncthreads(); }
  m = red[0]; __syncthreads();
  float sum = 0.0f;
  for (int i = t; i < T_; i += 256) { float e = __expf(p[i] - m); p[i] = e; sum += e; }
  red[t] = sum; __syncthreads();
  for (int s = 128; s > 0; s >>= 1) { if (t < s) red[t] += red[t + s]; __syncthreads(); }
  float inv = 1.0f / red[0];
  for (int i = t; i < T_; i += 256) p[i] *= inv;
}

// ---------------- y = att @ V  (A f32 converted in-register, V pre-transposed [DH,T] f16)
// Wave tile 32x64 (full DH); V fragments reused across the two row fragments.
__global__ void __launch_bounds__(256)
attn_av(const float* __restrict__ att, const half_t* __restrict__ vT,
        half_t* __restrict__ y16)
{
  int lane = threadIdx.x & 31;
  int wave = threadIdx.x >> 5;
  int bh = blockIdx.y;
  int b = bh >> 4, h = bh & 15;
  int qrow0 = (blockIdx.x * 8 + wave) * 32;
  const float*  S  = att + (size_t)bh * T_ * T_;
  const half_t* Vt = vT  + (size_t)bh * DH_ * T_;

  v8f acc[2][4] = {};
  int kend = qrow0 + 32;   // causal bound (multiple of 32; zeros beyond diagonal harmless)
  for (int k0 = 0; k0 < kend; k0 += 32) {
    v16h a0 = load_frag_f32(S, T_, qrow0,      k0, lane);
    v16h a1 = load_frag_f32(S, T_, qrow0 + 16, k0, lane);
#pragma unroll
    for (int j = 0; j < 4; ++j) {
      v16h bm = load_frag_f16(Vt, T_, 16 * j, k0, lane);
      acc[0][j] = __builtin_amdgcn_wmma_f32_16x16x32_f16(false, a0, false, bm,
                                                         (short)0, acc[0][j], false, false);
      acc[1][j] = __builtin_amdgcn_wmma_f32_16x16x32_f16(false, a1, false, bm,
                                                         (short)0, acc[1][j], false, false);
    }
  }
  int cn = lane & 15, rb = (lane >> 4) << 3;
#pragma unroll
  for (int j = 0; j < 4; ++j) {
    int d = 16 * j + cn;
#pragma unroll
    for (int i = 0; i < 2; ++i) {
#pragma unroll
      for (int r = 0; r < 8; ++r) {
        int row = qrow0 + 16 * i + rb + r;
        y16[((size_t)(b * T_ + row)) * C_ + h * DH_ + d] = (half_t)acc[i][j][r];
      }
    }
  }
}

// ---------------- LayerNorm fused with f32 -> f16 cast
__global__ void __launch_bounds__(256)
ln_cast(const float* __restrict__ x, const float* __restrict__ g,
        const float* __restrict__ beta, half_t* __restrict__ out16)
{
  __shared__ float red[256];
  const float* p = x + (size_t)blockIdx.x * C_;
  int t = threadIdx.x;
  float s = 0.0f;
  for (int i = t; i < C_; i += 256) s += p[i];
  red[t] = s; __syncthreads();
  for (int st = 128; st > 0; st >>= 1) { if (t < st) red[t] += red[t + st]; __syncthreads(); }
  float mu = red[0] * (1.0f / C_); __syncthreads();
  float v = 0.0f;
  for (int i = t; i < C_; i += 256) { float d = p[i] - mu; v += d * d; }
  red[t] = v; __syncthreads();
  for (int st = 128; st > 0; st >>= 1) { if (t < st) red[t] += red[t + st]; __syncthreads(); }
  float rs = rsqrtf(red[0] * (1.0f / C_) + 1e-5f);
  for (int i = t; i < C_; i += 256)
    out16[(size_t)blockIdx.x * C_ + i] = (half_t)((p[i] - mu) * rs * g[i] + beta[i]);
}

// ---------------- weight transpose + cast: W[K,N] f32 -> Wt[N,K] f16
__global__ void transpose_cast(const float* __restrict__ W, half_t* __restrict__ Wt,
                               int K, int N)
{
  size_t idx = (size_t)blockIdx.x * blockDim.x + threadIdx.x;
  if (idx >= (size_t)K * N) return;
  int k = (int)(idx / N), n = (int)(idx % N);
  Wt[(size_t)n * K + k] = (half_t)W[idx];
}

// ---------------- split qkv f32 [B,T,3C] -> q16/k16 [B,H,T,DH], vT16 [B,H,DH,T]
__global__ void split_qkv(const float* __restrict__ qkv,
                          half_t* __restrict__ q16, half_t* __restrict__ k16,
                          half_t* __restrict__ vT16)
{
  size_t idx = (size_t)blockIdx.x * blockDim.x + threadIdx.x;
  if (idx >= (size_t)B_ * T_ * C_) return;
  int c = (int)(idx % C_);
  size_t bt = idx / C_;
  int t = (int)(bt % T_);
  int b = (int)(bt / T_);
  int h = c / DH_, d = c % DH_;
  const float* base = qkv + bt * (3 * C_);
  size_t bh = (size_t)(b * H_ + h);
  q16 [(bh * T_ + t) * DH_ + d] = (half_t)base[c];
  k16 [(bh * T_ + t) * DH_ + d] = (half_t)base[C_ + c];
  vT16[(bh * DH_ + d) * T_ + t] = (half_t)base[2 * C_ + c];
}

extern "C" void kernel_launch(void* const* d_in, const int* in_sizes, int n_in,
                              void* d_out, int out_size, void* d_ws, size_t ws_size,
                              hipStream_t stream)
{
  const float* x      = (const float*)d_in[0];
  const float* W_attn = (const float*)d_in[1];
  const float* b_attn = (const float*)d_in[2];
  const float* W_o    = (const float*)d_in[3];
  const float* b_o    = (const float*)d_in[4];
  const float* W_fc   = (const float*)d_in[5];
  const float* b_fc   = (const float*)d_in[6];
  const float* W_fc2  = (const float*)d_in[7];
  const float* b_fc2  = (const float*)d_in[8];
  const float* g1     = (const float*)d_in[9];
  const float* beta1  = (const float*)d_in[10];
  const float* g2     = (const float*)d_in[11];
  const float* beta2  = (const float*)d_in[12];

  float* out_x   = (float*)d_out;                         // [B,T,C]
  float* out_att = out_x + (size_t)B_ * T_ * C_;          // [B,H,T,T]

  char* ws = (char*)d_ws;
  size_t off = 0;
  auto alloc = [&](size_t bytes) -> char* {
    char* p = ws + off;
    off += (bytes + 255) & ~(size_t)255;
    return p;
  };

  half_t* WattnT = (half_t*)alloc((size_t)3 * C_ * C_ * 2);     // [3C, C]
  half_t* WoT    = (half_t*)alloc((size_t)C_ * C_ * 2);         // [C, C]
  half_t* WfcT   = (half_t*)alloc((size_t)4 * C_ * C_ * 2);     // [4C, C]
  half_t* Wfc2T  = (half_t*)alloc((size_t)C_ * 4 * C_ * 2);     // [C, 4C]
  half_t* h16    = (half_t*)alloc((size_t)MTOK * C_ * 2);
  float*  qkv32  = (float*) alloc((size_t)MTOK * 3 * C_ * 4);
  half_t* q16    = (half_t*)alloc((size_t)B_ * H_ * T_ * DH_ * 2);
  half_t* k16    = (half_t*)alloc((size_t)B_ * H_ * T_ * DH_ * 2);
  half_t* vT16   = (half_t*)alloc((size_t)B_ * H_ * DH_ * T_ * 2);
  half_t* y16    = (half_t*)alloc((size_t)MTOK * C_ * 2);
  float*  x1     = (float*) alloc((size_t)MTOK * C_ * 4);
  half_t* h2_16  = (half_t*)alloc((size_t)MTOK * C_ * 2);
  half_t* m16    = (half_t*)alloc((size_t)MTOK * 4 * C_ * 2);

  // 1) weights -> transposed f16
  {
    size_t n;
    n = (size_t)C_ * 3 * C_;
    transpose_cast<<<dim3((n + 255) / 256), 256, 0, stream>>>(W_attn, WattnT, C_, 3 * C_);
    n = (size_t)C_ * C_;
    transpose_cast<<<dim3((n + 255) / 256), 256, 0, stream>>>(W_o, WoT, C_, C_);
    n = (size_t)C_ * 4 * C_;
    transpose_cast<<<dim3((n + 255) / 256), 256, 0, stream>>>(W_fc, WfcT, C_, 4 * C_);
    n = (size_t)4 * C_ * C_;
    transpose_cast<<<dim3((n + 255) / 256), 256, 0, stream>>>(W_fc2, Wfc2T, 4 * C_, C_);
  }

  // 2) LN1 -> h16
  ln_cast<<<MTOK, 256, 0, stream>>>(x, g1, beta1, h16);

  // 3) qkv = h16 @ W_attn + b_attn        [4096 x 3072]
  wmma_gemm_bias<<<dim3(MTOK / 64, (3 * C_) / 256), 256, 0, stream>>>(
      h16, WattnT, b_attn, nullptr, qkv32, nullptr, MTOK, 3 * C_, C_, 0);

  // 4) split + cast + V transpose
  split_qkv<<<dim3(((size_t)B_ * T_ * C_ + 255) / 256), 256, 0, stream>>>(
      qkv32, q16, k16, vT16);

  // 5) scores -> att (masked, scaled)
  attn_scores<<<dim3(T_ / 64, T_ / 256, B_ * H_), 256, 0, stream>>>(q16, k16, out_att);

  // 6) softmax rows in place
  softmax_rows<<<dim3(B_ * H_ * T_), 256, 0, stream>>>(out_att);

  // 7) y = att @ V  -> y16 in [B,T,C] layout
  attn_av<<<dim3(T_ / 256, B_ * H_), 256, 0, stream>>>(out_att, vT16, y16);

  // 8) x1 = x + y @ W_o + b_o
  wmma_gemm_bias<<<dim3(MTOK / 64, C_ / 256), 256, 0, stream>>>(
      y16, WoT, b_o, x, x1, nullptr, MTOK, C_, C_, 0);

  // 9) LN2 -> h2_16
  ln_cast<<<MTOK, 256, 0, stream>>>(x1, g2, beta2, h2_16);

  // 10) m16 = gelu(h2 @ W_fc + b_fc)      [4096 x 4096], f16 out
  wmma_gemm_bias<<<dim3(MTOK / 64, (4 * C_) / 256), 256, 0, stream>>>(
      h2_16, WfcT, b_fc, nullptr, nullptr, m16, MTOK, 4 * C_, C_, 1);

  // 11) out_x = x1 + m16 @ W_fc2 + b_fc2  [4096 x 1024], K = 4096
  wmma_gemm_bias<<<dim3(MTOK / 64, C_ / 256), 256, 0, stream>>>(
      m16, Wfc2T, b_fc2, x1, out_x, nullptr, MTOK, C_, 4 * C_, 0);

  (void)in_sizes; (void)n_in; (void)out_size; (void)ws_size;
}
